// NodeAttentionPool_11029476016738
// MI455X (gfx1250) — compile-verified, matrix operand order checked by reading, not run on hardware
//
#include <hip/hip_runtime.h>

typedef __attribute__((ext_vector_type(4)))  float   f4;
typedef __attribute__((ext_vector_type(8)))  float   v8f;
typedef __attribute__((ext_vector_type(8)))  __bf16  v8bf;
typedef __attribute__((ext_vector_type(16))) __bf16  v16bf;

#define WSTRIDE 136   // 128 bf16 + 8 pad (4-dword) per row: spreads LDS banks

__device__ __forceinline__ v16bf cat8(v8bf a, v8bf b) {
    return __builtin_shufflevector(a, b, 0,1,2,3,4,5,6,7,8,9,10,11,12,13,14,15);
}

// Branch-free tanh: native v_tanh_f32 if the toolchain exposes it, else
// 1 - 2*rcp(exp(2z)+1)  (exact at +-inf: exp->inf => 1, exp->0 => -1).
__device__ __forceinline__ float fast_tanh(float z) {
#if __has_builtin(__builtin_amdgcn_tanhf)
    return __builtin_amdgcn_tanhf(z);
#else
    float t = __expf(2.0f * z);
    return 1.0f - 2.0f * __builtin_amdgcn_rcpf(t + 1.0f);
#endif
}

// ---------------------------------------------------------------------------
// Kernel 1: s[i] = score_w . tanh(x[i] @ proj_w^T + proj_b) + score_b
// One wave per PAIR of 16-node row tiles (T=2: each B fragment read from LDS
// feeds 6 WMMAs). proj_w staged per block in LDS as bf16 hi/lo (3-term split).
// LDS offsets laundered via empty asm so LICM cannot hoist B loads (-> no
// register spill of the 64KB weight set).
// ---------------------------------------------------------------------------
__launch_bounds__(256)
__global__ void score_kernel(const float* __restrict__ x,
                             const float* __restrict__ proj_w,
                             const float* __restrict__ proj_b,
                             const float* __restrict__ score_w,
                             const float* __restrict__ score_b,
                             float* __restrict__ s_out, int N) {
    __shared__ __bf16 lwhi[128 * WSTRIDE];
    __shared__ __bf16 lwlo[128 * WSTRIDE];
    __shared__ float  lbias[128];
    __shared__ float  lsw[128];

    const int tid  = threadIdx.x;
    const int warp = tid >> 5;
    const int lane = tid & 31;
    const int half = lane >> 4;     // K-interleave half (ISA 16-bit A/B layout)
    const int l16  = lane & 15;

    // ---- stage proj_w -> LDS bf16 hi/lo, plus bias / score_w (once/block) ----
    for (int e = tid; e < 128 * 128; e += 256) {
        int r = e >> 7, cc = e & 127;
        float f = proj_w[e];
        __bf16 h = (__bf16)f;
        lwhi[r * WSTRIDE + cc] = h;
        lwlo[r * WSTRIDE + cc] = (__bf16)(f - (float)h);
    }
    if (tid < 128) { lbias[tid] = proj_b[tid]; lsw[tid] = score_w[tid]; }
    __syncthreads();

    const float sb = score_b[0];
    const int numTiles  = (N + 15) >> 4;
    const int numPairs  = (numTiles + 1) >> 1;
    const int waveId    = blockIdx.x * 8 + warp;
    const int waveCount = gridDim.x * 8;

    for (int p = waveId; p < numPairs; p += waveCount) {
        const int ta = 2 * p;
        const int tb = 2 * p + 1;

        int rowA = ta * 16 + l16; if (rowA >= N) rowA = N - 1;
        int rowB = tb * 16 + l16; if (rowB >= N) rowB = N - 1;
        const float* pxA = x + (long long)rowA * 128;
        const float* pxB = x + (long long)rowB * 128;

        // ---- A fragments (hi/lo) for both tiles, all 4 K-chunks, ONCE ----
        v16bf ahiA[4], aloA[4], ahiB[4], aloB[4];
#pragma unroll
        for (int c = 0; c < 4; ++c) {
            const int s0 = 32 * c + half * 8;
            const int s1 = s0 + 16;
#pragma unroll
            for (int m = 0; m < 2; ++m) {
                const float* px = m ? pxB : pxA;
                f4 t0 = *(const f4*)(px + s0);
                f4 t1 = *(const f4*)(px + s0 + 4);
                f4 t2 = *(const f4*)(px + s1);
                f4 t3 = *(const f4*)(px + s1 + 4);
                float vals[16];
#pragma unroll
                for (int j = 0; j < 4; ++j) {
                    vals[j] = t0[j]; vals[4 + j] = t1[j];
                    vals[8 + j] = t2[j]; vals[12 + j] = t3[j];
                }
                v16bf hi, lo;
#pragma unroll
                for (int j = 0; j < 16; ++j) {
                    __bf16 h = (__bf16)vals[j];
                    hi[j] = h;
                    lo[j] = (__bf16)(vals[j] - (float)h);
                }
                if (m) { ahiB[c] = hi; aloB[c] = lo; }
                else   { ahiA[c] = hi; aloA[c] = lo; }
            }
        }

        float saccA[8], saccB[8];
#pragma unroll
        for (int r = 0; r < 8; ++r) { saccA[r] = 0.0f; saccB[r] = 0.0f; }

        // ---- sweep 8 column slabs; each B fragment feeds both tiles ----
#pragma clang loop unroll(disable)
        for (int j = 0; j < 8; ++j) {
            int jb = (j * 16 + l16) * WSTRIDE;
            asm volatile("" : "+v"(jb));   // opaque: forbid LICM hoist of B loads
            const __bf16* rh = &lwhi[jb];
            const __bf16* rl = &lwlo[jb];

            v8f accA = {}, accB = {};
#pragma unroll
            for (int c = 0; c < 4; ++c) {
                const int s0 = 32 * c + half * 8;
                const int s1 = s0 + 16;
                v16bf bhi = cat8(*(const v8bf*)(rh + s0), *(const v8bf*)(rh + s1));
                v16bf blo = cat8(*(const v8bf*)(rl + s0), *(const v8bf*)(rl + s1));
                accA = __builtin_amdgcn_wmma_f32_16x16x32_bf16(false, ahiA[c], false, bhi,
                                                               (short)0, accA, false, false);
                accA = __builtin_amdgcn_wmma_f32_16x16x32_bf16(false, ahiA[c], false, blo,
                                                               (short)0, accA, false, false);
                accA = __builtin_amdgcn_wmma_f32_16x16x32_bf16(false, aloA[c], false, bhi,
                                                               (short)0, accA, false, false);
                accB = __builtin_amdgcn_wmma_f32_16x16x32_bf16(false, ahiB[c], false, bhi,
                                                               (short)0, accB, false, false);
                accB = __builtin_amdgcn_wmma_f32_16x16x32_bf16(false, ahiB[c], false, blo,
                                                               (short)0, accB, false, false);
                accB = __builtin_amdgcn_wmma_f32_16x16x32_bf16(false, aloB[c], false, bhi,
                                                               (short)0, accB, false, false);
            }
            // C/D layout: VGPR r -> M=r (lanes 0-15) / M=r+8 (lanes 16-31); lane%16 = col
            const float bj  = lbias[j * 16 + l16];
            const float swj = lsw[j * 16 + l16];
#pragma unroll
            for (int r = 0; r < 8; ++r) {
                saccA[r] += fast_tanh(accA[r] + bj) * swj;
                saccB[r] += fast_tanh(accB[r] + bj) * swj;
            }
        }

        // ---- reduce score partials over the 16 lanes of each half; write s ----
#pragma unroll
        for (int r = 0; r < 8; ++r) {
            float totA = saccA[r], totB = saccB[r];
#pragma unroll
            for (int m = 8; m >= 1; m >>= 1) {
                totA += __shfl_xor(totA, m, 32);
                totB += __shfl_xor(totB, m, 32);
            }
            if (l16 == 0) {
                int roA = ta * 16 + r + half * 8;
                int roB = tb * 16 + r + half * 8;
                if (roA < N) s_out[roA] = totA + sb;
                if (roB < N) s_out[roB] = totB + sb;
            }
        }
    }
}

// ---------------------------------------------------------------------------
// Kernel 2: segment softmax over sorted `batch` + weighted scatter-sum of x.
// ---------------------------------------------------------------------------
__device__ __forceinline__ int lbound(const int* __restrict__ a, int n, int key) {
    int lo = 0, hi = n;
    while (lo < hi) { int mid = (lo + hi) >> 1; if (a[mid] < key) lo = mid + 1; else hi = mid; }
    return lo;
}

__launch_bounds__(256)
__global__ void pool_kernel(const float* __restrict__ x, const float* __restrict__ s,
                            const int* __restrict__ batch, float* __restrict__ out, int N) {
    __shared__ int   seg[2];
    __shared__ float red[256];
    __shared__ float mv_s, inv_s;
    const int g = blockIdx.x;
    const int tid = threadIdx.x;

    if (tid == 0) seg[0] = lbound(batch, N, g);
    if (tid == 1) seg[1] = lbound(batch, N, g + 1);
    __syncthreads();
    const int s0 = seg[0], s1 = seg[1];

    // pass 1: segment max
    float lm = -3.4e38f;
    for (int i = s0 + tid; i < s1; i += 256) lm = fmaxf(lm, s[i]);
    red[tid] = lm; __syncthreads();
    for (int off = 128; off; off >>= 1) {
        if (tid < off) red[tid] = fmaxf(red[tid], red[tid + off]);
        __syncthreads();
    }
    if (tid == 0) mv_s = (red[0] > -3.0e38f) ? red[0] : 0.0f;  // empty-segment guard
    __syncthreads();
    const float mv = mv_s;

    // pass 2: denom
    float le = 0.0f;
    for (int i = s0 + tid; i < s1; i += 256) le += __expf(s[i] - mv);
    red[tid] = le; __syncthreads();
    for (int off = 128; off; off >>= 1) {
        if (tid < off) red[tid] += red[tid + off];
        __syncthreads();
    }
    if (tid == 0) inv_s = 1.0f / fmaxf(red[0], 1e-30f);
    __syncthreads();
    const float inv = inv_s;

    // pass 3: weighted column-parallel accumulation (2 nodes/iter, 128 cols)
    const int sub = tid >> 7;
    const int c   = tid & 127;
    float acc = 0.0f;
    for (int i = s0 + sub; i < s1; i += 2) {
        float w = __expf(s[i] - mv) * inv;
        acc += x[(long long)i * 128 + c] * w;
    }
    red[tid] = acc; __syncthreads();
    if (tid < 128) out[(long long)g * 128 + tid] = red[tid] + red[tid + 128];
}

// ---------------------------------------------------------------------------
extern "C" void kernel_launch(void* const* d_in, const int* in_sizes, int n_in,
                              void* d_out, int out_size, void* d_ws, size_t ws_size,
                              hipStream_t stream) {
    const float* x       = (const float*)d_in[0];
    const float* proj_w  = (const float*)d_in[1];
    const float* proj_b  = (const float*)d_in[2];
    const float* score_w = (const float*)d_in[3];
    const float* score_b = (const float*)d_in[4];
    const int*   batch   = (const int*)d_in[5];

    const int N = in_sizes[5];          // 500000
    const int C = in_sizes[2];          // 128
    const int G = out_size / C;         // 4096

    float* s_buf = (float*)d_ws;        // workspace: s[N] fp32

    int numTiles = (N + 15) / 16;
    int numPairs = (numTiles + 1) / 2;
    int blocks = (numPairs + 7) / 8;
    if (blocks > 512) blocks = 512;
    if (blocks < 1) blocks = 1;
    score_kernel<<<blocks, 256, 0, stream>>>(x, proj_w, proj_b, score_w, score_b,
                                             s_buf, N);

    pool_kernel<<<G, 256, 0, stream>>>(x, s_buf, batch, (float*)d_out, N);
}
